// ElasticMultiheadAttention_83013127897754
// MI455X (gfx1250) — matrix operand order repaired and checked by reading, not compiled, not run on hardware
//
#include <hip/hip_runtime.h>
#include <hip/hip_bf16.h>

// ---------------------------------------------------------------------------
// MHA forward for MI455X (gfx1250, wave32, WMMA).
// bf16 v_wmma_f32_16x16x32_bf16 everywhere (fp32 accumulate), fused
// score+softmax strip in LDS, attn matrix written to HBM exactly once.
// fp32->bf16 via native converts (fptrunc), prefetch for K-loop streaming.
// ---------------------------------------------------------------------------

constexpr int Bn  = 2;
constexpr int Sn  = 2048;
constexpr int Dn  = 1024;
constexpr int Hn  = 16;
constexpr int HDn = 64;          // head dim
constexpr int Mn  = Bn * Sn;     // 4096 token rows

typedef __attribute__((ext_vector_type(16))) __bf16          bf16x16;
typedef __attribute__((ext_vector_type(8)))  float           f32x8;
typedef __attribute__((ext_vector_type(16))) unsigned short  u16x16;

__device__ __forceinline__ unsigned short f2bf_bits(float f) {
  return __builtin_bit_cast(unsigned short, (__bf16)f);   // native v_cvt RNE
}

__device__ __forceinline__ f32x8 wmma_bf16(bf16x16 a, bf16x16 b, f32x8 c) {
  // 8 args: (neg_a, A, neg_b, B, c_mod, C, reuse_a, reuse_b)
  return __builtin_amdgcn_wmma_f32_16x16x32_bf16(false, a, false, b,
                                                 (short)0, c, false, false);
}

// ---- A fragment (16x32, M x K), fp32 source row-major, convert to bf16 ----
// Lane l: m = l&15. Lane group 0 holds K {0..15}, group 1 {16..31}.
__device__ __forceinline__ bf16x16 load_a_f32(const float* __restrict__ A,
                                              int lda, int k0) {
  int l = threadIdx.x & 31, m = l & 15, hi = l >> 4;
  const float* p0 = A + (size_t)m * lda + k0 + hi * 16;
  const float* p1 = p0 + 8;
  bf16x16 v;
#pragma unroll
  for (int j = 0; j < 8; ++j) { v[j] = (__bf16)p0[j]; v[j + 8] = (__bf16)p1[j]; }
  return v;
}

__device__ __forceinline__ bf16x16 load_a_bf16(const unsigned short* __restrict__ A,
                                               int lda, int k0) {
  int l = threadIdx.x & 31, m = l & 15, hi = l >> 4;
  const unsigned short* p0 = A + (size_t)m * lda + k0 + hi * 16;
  const unsigned short* p1 = p0 + 8;
  u16x16 u;
#pragma unroll
  for (int j = 0; j < 8; ++j) { u[j] = p0[j]; u[j + 8] = p1[j]; }
  return __builtin_bit_cast(bf16x16, u);
}

// ---- B fragment (32x16, K x N) where column n of B is a CONTIGUOUS row  ----
// of the source matrix (i.e. B[k][n] = Src[n][k]).  Lane l: n = l&15,
// elements j = Src[n][k0 + (l>>4)*16 + j].
__device__ __forceinline__ bf16x16 load_b_rows_f32(const float* __restrict__ Bm,
                                                   int ldb, int k0) {
  int l = threadIdx.x & 31, n = l & 15, hi = l >> 4;
  const float* p = Bm + (size_t)n * ldb + k0 + hi * 16;
  bf16x16 v;
#pragma unroll
  for (int j = 0; j < 16; ++j) v[j] = (__bf16)p[j];
  return v;
}

__device__ __forceinline__ bf16x16 load_b_rows_bf16(const unsigned short* __restrict__ Bm,
                                                    int ldb, int k0) {
  int l = threadIdx.x & 31, n = l & 15, hi = l >> 4;
  const unsigned short* p = Bm + (size_t)n * ldb + k0 + hi * 16;
  u16x16 u;
#pragma unroll
  for (int j = 0; j < 16; ++j) u[j] = p[j];
  return __builtin_bit_cast(bf16x16, u);
}

// ---------------------------------------------------------------------------
// Kernel 1: QKV projection.  Y[m][n] = sum_k X[m][k]*W[n][k] + bias[n].
// MODE 0: store bf16 head layout [b][h][s][hd]   (Q and K)
// MODE 1: store bf16 transposed  [b][h][hd][s]   (V, so PV GEMM B is contiguous)
// Block = 8 waves; wave w does a 16x64 C tile (4 WMMA accumulators, shared A).
// ---------------------------------------------------------------------------
template <int MODE>
__global__ __launch_bounds__(256) void proj_kernel(
    const float* __restrict__ X, const float* __restrict__ W,
    const float* __restrict__ bias, unsigned short* __restrict__ Out) {
  int wave = threadIdx.x >> 5;
  int m0 = blockIdx.x * 16;
  int n0 = blockIdx.y * 512 + wave * 64;
  int l = threadIdx.x & 31, hi = l >> 4, nl = l & 15;

  const float* Arow = X + (size_t)m0 * Dn + (size_t)(l & 15) * Dn;

  f32x8 acc[4] = {};
  for (int k0 = 0; k0 < Dn; k0 += 32) {
    __builtin_prefetch(Arow + k0 + 64, 0, 1);           // global_prefetch_b8
    bf16x16 a = load_a_f32(X + (size_t)m0 * Dn, Dn, k0);
#pragma unroll
    for (int t = 0; t < 4; ++t) {
      bf16x16 b = load_b_rows_f32(W + (size_t)(n0 + t * 16) * Dn, Dn, k0);
      acc[t] = wmma_bf16(a, b, acc[t]);
    }
  }

#pragma unroll
  for (int t = 0; t < 4; ++t) {
    int n = n0 + t * 16 + nl;
    int h = n / HDn, hd = n % HDn;
    float bv = bias[n];
#pragma unroll
    for (int r = 0; r < 8; ++r) {
      int m = m0 + r + 8 * hi;
      int bb = m / Sn, s = m % Sn;
      unsigned short o = f2bf_bits(acc[t][r] + bv);
      if (MODE == 0)
        Out[(((size_t)(bb * Hn + h) * Sn + s) * HDn) + hd] = o;
      else
        Out[(((size_t)(bb * Hn + h) * HDn + hd) * Sn) + s] = o;
    }
  }
}

// ---------------------------------------------------------------------------
// Kernel 2: scores = (Q Kt)/8, fused row softmax.  One block = 16 query rows
// for one (b,h); the full 16x2048 fp32 score strip lives in dynamic LDS
// (128 KB of the 320 KB WGP pool).  fp32 attn written to d_out once.
// ---------------------------------------------------------------------------
__global__ __launch_bounds__(256) void scores_softmax_kernel(
    const unsigned short* __restrict__ Qb, const unsigned short* __restrict__ Kb,
    float* __restrict__ attn) {
  extern __shared__ float sm[];               // [16][Sn]
  int qt = blockIdx.x, h = blockIdx.y, b = blockIdx.z;
  int wave = threadIdx.x >> 5;
  int l = threadIdx.x & 31, hi = l >> 4, nl = l & 15;

  const unsigned short* Q = Qb + ((size_t)(b * Hn + h) * Sn + qt * 16) * HDn;
  const unsigned short* Km = Kb + (size_t)(b * Hn + h) * Sn * HDn;

  bf16x16 a0 = load_a_bf16(Q, HDn, 0);
  bf16x16 a1 = load_a_bf16(Q, HDn, 32);

  for (int t = wave; t < Sn / 16; t += 8) {   // 16 key tiles per wave
    __builtin_prefetch(Km + (size_t)(t + 8) * 16 * HDn + (size_t)nl * HDn, 0, 1);
    f32x8 acc = {};
    bf16x16 b0 = load_b_rows_bf16(Km + (size_t)t * 16 * HDn, HDn, 0);
    bf16x16 b1 = load_b_rows_bf16(Km + (size_t)t * 16 * HDn, HDn, 32);
    acc = wmma_bf16(a0, b0, acc);
    acc = wmma_bf16(a1, b1, acc);
#pragma unroll
    for (int r = 0; r < 8; ++r)
      sm[(size_t)(r + 8 * hi) * Sn + t * 16 + nl] = acc[r] * 0.125f;
  }
  __syncthreads();

  // Row softmax: wave handles rows 2*wave, 2*wave+1; wave32 shuffle reductions.
  float* outbase = attn + ((size_t)(b * Hn + h) * Sn + (size_t)qt * 16) * Sn;
  for (int rr = 0; rr < 2; ++rr) {
    int row = wave * 2 + rr;
    float* srow = sm + (size_t)row * Sn;
    float mx = -1e30f;
    for (int i = l; i < Sn; i += 32) mx = fmaxf(mx, srow[i]);
#pragma unroll
    for (int off = 16; off > 0; off >>= 1) mx = fmaxf(mx, __shfl_xor(mx, off, 32));
    float sum = 0.f;
    for (int i = l; i < Sn; i += 32) {
      float e = __expf(srow[i] - mx);
      srow[i] = e;
      sum += e;
    }
#pragma unroll
    for (int off = 16; off > 0; off >>= 1) sum += __shfl_xor(sum, off, 32);
    float inv = 1.0f / sum;
    for (int i = l; i < Sn; i += 32) outbase[(size_t)row * Sn + i] = srow[i] * inv;
  }
}

// ---------------------------------------------------------------------------
// Kernel 3: attn_output = P @ V.  Per (b,h): M=Sn, N=HDn, K=Sn.
// A = fp32 attn (bf16-converted on load), B = transposed bf16 V (contiguous).
// Wave = 16x64 tile (entire N).  Output bf16 in flat [b*S+s][h*HD+hd].
// ---------------------------------------------------------------------------
__global__ __launch_bounds__(256) void attn_v_kernel(
    const float* __restrict__ P, const unsigned short* __restrict__ Vt,
    unsigned short* __restrict__ Ob) {
  int wave = threadIdx.x >> 5;
  int h = blockIdx.y, b = blockIdx.z;
  int m0 = blockIdx.x * 128 + wave * 16;
  int l = threadIdx.x & 31, hi = l >> 4, nl = l & 15;

  const float* Pm = P + ((size_t)(b * Hn + h) * Sn + m0) * Sn;
  const unsigned short* Vh = Vt + (size_t)(b * Hn + h) * HDn * Sn;
  const float* Arow = Pm + (size_t)(l & 15) * Sn;

  f32x8 acc[4] = {};
  for (int k0 = 0; k0 < Sn; k0 += 32) {
    __builtin_prefetch(Arow + k0 + 64, 0, 1);           // stream next A lines
    bf16x16 a = load_a_f32(Pm, Sn, k0);
#pragma unroll
    for (int t = 0; t < 4; ++t) {
      bf16x16 bb = load_b_rows_bf16(Vh + (size_t)(t * 16) * Sn, Sn, k0);
      acc[t] = wmma_bf16(a, bb, acc[t]);
    }
  }

#pragma unroll
  for (int t = 0; t < 4; ++t) {
    int hd = t * 16 + nl;
#pragma unroll
    for (int r = 0; r < 8; ++r) {
      int s = m0 + r + 8 * hi;
      Ob[(size_t)(b * Sn + s) * Dn + h * HDn + hd] = f2bf_bits(acc[t][r]);
    }
  }
}

// ---------------------------------------------------------------------------
// Kernel 4: output = attn_out @ Wo^T + bo  (fp32 result to d_out).
// ---------------------------------------------------------------------------
__global__ __launch_bounds__(256) void out_proj_kernel(
    const unsigned short* __restrict__ Ab, const float* __restrict__ Wo,
    const float* __restrict__ bo, float* __restrict__ Y) {
  int wave = threadIdx.x >> 5;
  int m0 = blockIdx.x * 16;
  int n0 = blockIdx.y * 512 + wave * 64;
  int l = threadIdx.x & 31, hi = l >> 4, nl = l & 15;

  f32x8 acc[4] = {};
  for (int k0 = 0; k0 < Dn; k0 += 32) {
    bf16x16 a = load_a_bf16(Ab + (size_t)m0 * Dn, Dn, k0);
#pragma unroll
    for (int t = 0; t < 4; ++t) {
      bf16x16 bb = load_b_rows_f32(Wo + (size_t)(n0 + t * 16) * Dn, Dn, k0);
      acc[t] = wmma_bf16(a, bb, acc[t]);
    }
  }

#pragma unroll
  for (int t = 0; t < 4; ++t) {
    int n = n0 + t * 16 + nl;
    float bv = bo[n];
#pragma unroll
    for (int r = 0; r < 8; ++r) {
      int m = m0 + r + 8 * hi;
      Y[(size_t)m * Dn + n] = acc[t][r] + bv;
    }
  }
}

// ---------------------------------------------------------------------------
extern "C" void kernel_launch(void* const* d_in, const int* in_sizes, int n_in,
                              void* d_out, int out_size, void* d_ws, size_t ws_size,
                              hipStream_t stream) {
  const float* query = (const float*)d_in[0];
  const float* key   = (const float*)d_in[1];
  const float* value = (const float*)d_in[2];
  const float* wq = (const float*)d_in[3];
  const float* bq = (const float*)d_in[4];
  const float* wk = (const float*)d_in[5];
  const float* bk = (const float*)d_in[6];
  const float* wv = (const float*)d_in[7];
  const float* bv = (const float*)d_in[8];
  const float* wo = (const float*)d_in[9];
  const float* bo = (const float*)d_in[10];

  float* out0 = (float*)d_out;                          // (B,S,D)
  float* attn = out0 + (size_t)Bn * Sn * Dn;            // (B,H,S,S)

  // Workspace: four bf16 tensors of M*D elements (8 MB each, 32 MB total).
  const size_t tsz = (size_t)Mn * Dn;                   // elements
  unsigned short* q_bf = (unsigned short*)d_ws;
  unsigned short* k_bf = q_bf + tsz;
  unsigned short* v_t  = k_bf + tsz;                    // [b][h][hd][s]
  unsigned short* a_bf = v_t + tsz;                     // attn_output bf16

  dim3 projGrid(Mn / 16, Dn / 512);                     // 256 x 2 blocks
  proj_kernel<0><<<projGrid, 256, 0, stream>>>(query, wq, bq, q_bf);
  proj_kernel<0><<<projGrid, 256, 0, stream>>>(key,   wk, bk, k_bf);
  proj_kernel<1><<<projGrid, 256, 0, stream>>>(value, wv, bv, v_t);

  dim3 scoreGrid(Sn / 16, Hn, Bn);                      // 128 x 16 x 2
  size_t lds_bytes = (size_t)16 * Sn * sizeof(float);   // 128 KB dynamic LDS
  scores_softmax_kernel<<<scoreGrid, 256, lds_bytes, stream>>>(q_bf, k_bf, attn);

  dim3 pvGrid(Sn / 128, Hn, Bn);                        // 16 x 16 x 2
  attn_v_kernel<<<pvGrid, 256, 0, stream>>>(attn, v_t, a_bf);

  out_proj_kernel<<<projGrid, 256, 0, stream>>>(a_bf, wo, bo, out0);
}
